// MixtureOfExpertsLayer_19533511262530
// MI455X (gfx1250) — matrix operand order repaired and checked by reading, not compiled, not run on hardware
//
#include <hip/hip_runtime.h>
#include <math.h>
#include <stdint.h>

typedef __bf16 bf16;
typedef __attribute__((ext_vector_type(16))) __bf16 v16bf;
typedef __attribute__((ext_vector_type(8)))  __bf16 v8bf;
typedef __attribute__((ext_vector_type(8)))  float  v8f;

#define T_TOK   4096
#define DIM     768
#define HID     2688
#define NEXP    8
#define NASSIGN 8192
#define ZCOEFF  1e-5f

// ---------------------------------------------------------------- utilities

__global__ void moe_zero_kernel(int* __restrict__ counts) {
    if (threadIdx.x < NEXP) counts[threadIdx.x] = 0;
}

__device__ __forceinline__ void moe_wait_async0() {
#if defined(__has_builtin)
#if __has_builtin(__builtin_amdgcn_s_wait_asynccnt)
    __builtin_amdgcn_s_wait_asynccnt(0);
    return;
#else
    asm volatile("s_wait_asynccnt 0x0" ::: "memory");
    return;
#endif
#else
    asm volatile("s_wait_asynccnt 0x0" ::: "memory");
#endif
}

// Stage a 128x32 bf16 tile (row stride `ld` elements in global) into LDS via
// the CDNA5 async LDS-DMA path (GLOBAL_LOAD_ASYNC_TO_LDS_B128, ASYNCcnt).
// Rows >= rows_left are skipped (caller pre-zeroes the tile for partial M).
__device__ __forceinline__ void moe_async_tile_a(bf16* lds_tile,
                                                 const bf16* gbase, int ld,
                                                 int tid, int rows_left) {
    unsigned lbase = (unsigned)(uintptr_t)lds_tile;  // low 32 bits = LDS offset
#pragma unroll
    for (int it = 0; it < 2; ++it) {
        int c = tid + it * 256;            // 512 chunks of 16B (128 rows x 4)
        int mi = c >> 2, seg = c & 3;
        if (mi < rows_left) {
            unsigned lofs = lbase + (unsigned)(mi * 64 + seg * 16);
            unsigned gofs = (unsigned)((mi * ld + seg * 8) * 2);
            asm volatile("global_load_async_to_lds_b128 %0, %1, %2"
                         :: "v"(lofs), "v"(gofs), "s"(gbase)
                         : "memory");
        }
    }
}

// ---------------------------------------------------------------- router
// one wave (32 lanes) per token; gate_w (8x768 f32 = 24KB) cached in LDS.
__global__ __launch_bounds__(256)
void moe_router_kernel(const float* __restrict__ x,
                       const float* __restrict__ gate_w,
                       const float* __restrict__ expert_bias,
                       int* __restrict__ ridx, float* __restrict__ rw,
                       float* __restrict__ lse2, int* __restrict__ counts) {
    __shared__ float gws[NEXP * DIM];
    int tid = threadIdx.x;
    for (int i = tid; i < NEXP * DIM; i += 256) gws[i] = gate_w[i];
    __syncthreads();

    int wave = tid >> 5, lane = tid & 31;
    int t = blockIdx.x * 8 + wave;

    float acc[NEXP];
#pragma unroll
    for (int e = 0; e < NEXP; ++e) acc[e] = 0.0f;

    const float* xr = x + (size_t)t * DIM;
    for (int i = lane; i < DIM; i += 32) {
        float xv = xr[i];
#pragma unroll
        for (int e = 0; e < NEXP; ++e) acc[e] += xv * gws[e * DIM + i];
    }
#pragma unroll
    for (int e = 0; e < NEXP; ++e)
        for (int off = 16; off > 0; off >>= 1)
            acc[e] += __shfl_xor(acc[e], off, 32);

    if (lane == 0) {
        float bl[NEXP];
#pragma unroll
        for (int e = 0; e < NEXP; ++e) bl[e] = acc[e] + expert_bias[e];
        int i0 = 0;
#pragma unroll
        for (int e = 1; e < NEXP; ++e) if (bl[e] > bl[i0]) i0 = e;
        int i1 = (i0 == 0) ? 1 : 0;
#pragma unroll
        for (int e = 0; e < NEXP; ++e) if (e != i0 && bl[e] > bl[i1]) i1 = e;

        // softmax over the two selected UNbiased logits
        float s0 = acc[i0], s1 = acc[i1];
        float m2 = fmaxf(s0, s1);
        float e0 = expf(s0 - m2), e1 = expf(s1 - m2);
        float inv = 1.0f / (e0 + e1);

        // logsumexp over all 8 (for z-loss)
        float mx = acc[0];
#pragma unroll
        for (int e = 1; e < NEXP; ++e) mx = fmaxf(mx, acc[e]);
        float s = 0.0f;
#pragma unroll
        for (int e = 0; e < NEXP; ++e) s += expf(acc[e] - mx);
        float lse = mx + logf(s);
        lse2[t] = lse * lse;

        ridx[2 * t] = i0;  ridx[2 * t + 1] = i1;
        rw[2 * t] = e0 * inv;  rw[2 * t + 1] = e1 * inv;
        atomicAdd(&counts[i0], 1);
        atomicAdd(&counts[i1], 1);
    }
}

// ---------------------------------------------------------------- scan
// fixed-order z reduction + expert prefix sums (single block -> deterministic)
__global__ __launch_bounds__(256)
void moe_scan_kernel(const float* __restrict__ lse2, float* __restrict__ zscale,
                     const int* __restrict__ counts, int* __restrict__ offsets,
                     int* __restrict__ cursors) {
    __shared__ float red[256];
    int tid = threadIdx.x;
    float s = 0.0f;
    for (int t = tid; t < T_TOK; t += 256) s += lse2[t];
    red[tid] = s;
    __syncthreads();
    for (int w = 128; w > 0; w >>= 1) {
        if (tid < w) red[tid] += red[tid + w];
        __syncthreads();
    }
    if (tid == 0) {
        zscale[0] = ZCOEFF * red[0] / (float)T_TOK;
        int run = 0;
        for (int e = 0; e < NEXP; ++e) {
            offsets[e] = run;
            cursors[e] = run;
            run += counts[e];
        }
        offsets[NEXP] = run;
    }
}

// ---------------------------------------------------------------- gather
// one block per (token,k): claim a slot in the expert segment, cast row to bf16
__global__ __launch_bounds__(128)
void moe_gather_kernel(const float* __restrict__ x, const int* __restrict__ ridx,
                       int* __restrict__ cursors, int* __restrict__ pos_tk,
                       bf16* __restrict__ Xg) {
    int a = blockIdx.x;
    int t = a >> 1;
    __shared__ int spos;
    if (threadIdx.x == 0) {
        int e = ridx[a];
        int p = atomicAdd(&cursors[e], 1);
        spos = p;
        pos_tk[a] = p;
    }
    __syncthreads();
    int pos = spos;
    const float* xr = x + (size_t)t * DIM;
    bf16* dst = Xg + (size_t)pos * DIM;
    for (int d = threadIdx.x; d < DIM; d += 128) dst[d] = (bf16)xr[d];
}

// ---------------------------------------------------------------- WMMA frags
// A fragment 16x32 bf16 from row-major LDS tile (ld = 32 halfs).
// lane l: M = base + (l&15); h = l>>4; K = [h*8, h*8+8) and [16+h*8, 16+h*8+8)
__device__ inline v16bf moe_frag_a(const bf16* As, int mbase, int lane) {
    int m = mbase + (lane & 15);
    int h = lane >> 4;
    const bf16* p = As + m * 32 + h * 8;
    v8bf lo = *(const v8bf*)(p);
    v8bf hi = *(const v8bf*)(p + 16);
    v16bf r;
#pragma unroll
    for (int i = 0; i < 8; ++i) { r[i] = lo[i]; r[8 + i] = hi[i]; }
    return r;
}
// B fragment 32x16 bf16 from LDS tile stored TRANSPOSED as [n][k] (ld = 32).
// lane l: N = base + (l&15); h = l>>4; K = [h*8, h*8+8) and [16+h*8, 16+h*8+8)
__device__ inline v16bf moe_frag_b(const bf16* Bs, int nbase, int lane) {
    int n = nbase + (lane & 15);
    int h = lane >> 4;
    const bf16* p = Bs + n * 32 + h * 8;
    v8bf lo = *(const v8bf*)(p);
    v8bf hi = *(const v8bf*)(p + 16);
    v16bf r;
#pragma unroll
    for (int i = 0; i < 8; ++i) { r[i] = lo[i]; r[8 + i] = hi[i]; }
    return r;
}

// ---------------------------------------------------------------- GEMM1
// per-expert  silu(Xg @ Wg) * (Xg @ Wu) -> A2 (bf16).  128x128 tile, 8 waves.
__global__ __launch_bounds__(256)
void moe_gemm1_kernel(const bf16* __restrict__ Xg,
                      const float* __restrict__ w_gate,
                      const float* __restrict__ w_up,
                      const int* __restrict__ offsets,
                      bf16* __restrict__ A2) {
    int e = blockIdx.z;
    int seg0 = offsets[e];
    int nrows = offsets[e + 1] - seg0;
    int m0 = blockIdx.x * 128;
    if (m0 >= nrows) return;
    int n0 = blockIdx.y * 128;
    int rows_left = nrows - m0;

    __shared__ __align__(16) bf16 As[128 * 32];
    __shared__ __align__(16) bf16 Bg[128 * 32];  // transposed [n][k]
    __shared__ __align__(16) bf16 Bu[128 * 32];  // transposed [n][k]

    int tid = threadIdx.x;
    int lane = tid & 31, wave = tid >> 5;
    int wm = (wave & 3) * 32;   // 4 waves stacked in M
    int wn = (wave >> 2) * 64;  // 2 waves in N

    // partial M-tile: zero the A tile once; async loads only touch valid rows
    if (rows_left < 128) {
        for (int i = tid; i < 128 * 32; i += 256) As[i] = (bf16)0.0f;
    }

    v8f zero8 = {0.f, 0.f, 0.f, 0.f, 0.f, 0.f, 0.f, 0.f};
    v8f accg[2][4], accu[2][4];
#pragma unroll
    for (int a = 0; a < 2; ++a)
#pragma unroll
        for (int b = 0; b < 4; ++b) { accg[a][b] = zero8; accu[a][b] = zero8; }

    const float* Wg = w_gate + (size_t)e * DIM * HID;
    const float* Wu = w_up   + (size_t)e * DIM * HID;
    const bf16*  Ab = Xg + (size_t)(seg0 + m0) * DIM;

    for (int kk = 0; kk < DIM; kk += 32) {
        // A tile: async LDS DMA (no VGPR staging, ASYNCcnt-tracked)
        moe_async_tile_a(As, Ab + kk, DIM, tid, rows_left);
        // B tiles: fp32->bf16 convert + transpose into [n][k]
        for (int i = tid; i < 32 * 128; i += 256) {
            int ki = i >> 7, ni = i & 127;
            size_t gi = (size_t)(kk + ki) * HID + (size_t)(n0 + ni);
            Bg[ni * 32 + ki] = (bf16)Wg[gi];
            Bu[ni * 32 + ki] = (bf16)Wu[gi];
        }
        // prefetch next K-chunk of weights (global_prefetch_b8)
        if (kk + 32 < DIM) {
            int pk = tid >> 3, pn = (tid & 7) * 16;
            size_t pi = (size_t)(kk + 32 + pk) * HID + (size_t)(n0 + pn);
            __builtin_prefetch(&Wg[pi], 0, 1);
            __builtin_prefetch(&Wu[pi], 0, 1);
        }
        moe_wait_async0();
        __syncthreads();

        v16bf af0 = moe_frag_a(As, wm, lane);
        v16bf af1 = moe_frag_a(As, wm + 16, lane);
#pragma unroll
        for (int b = 0; b < 4; ++b) {
            v16bf bg = moe_frag_b(Bg, wn + b * 16, lane);
            v16bf bu = moe_frag_b(Bu, wn + b * 16, lane);
            accg[0][b] = __builtin_amdgcn_wmma_f32_16x16x32_bf16(
                false, af0, false, bg, (short)0, accg[0][b], false, false);
            accg[1][b] = __builtin_amdgcn_wmma_f32_16x16x32_bf16(
                false, af1, false, bg, (short)0, accg[1][b], false, false);
            accu[0][b] = __builtin_amdgcn_wmma_f32_16x16x32_bf16(
                false, af0, false, bu, (short)0, accu[0][b], false, false);
            accu[1][b] = __builtin_amdgcn_wmma_f32_16x16x32_bf16(
                false, af1, false, bu, (short)0, accu[1][b], false, false);
        }
        __syncthreads();
    }

    // epilogue: silu(g)*u -> bf16.  C layout: N = lane&15, M = (lane>>4)*8 + r
    int h = lane >> 4, nn = lane & 15;
#pragma unroll
    for (int a = 0; a < 2; ++a)
#pragma unroll
        for (int b = 0; b < 4; ++b)
#pragma unroll
            for (int r = 0; r < 8; ++r) {
                int m = wm + a * 16 + h * 8 + r;
                if (m < rows_left) {
                    int col = n0 + wn + b * 16 + nn;
                    float g = accg[a][b][r];
                    float u = accu[a][b][r];
                    float sg = g / (1.0f + expf(-g));
                    A2[(size_t)(seg0 + m0 + m) * HID + col] = (bf16)(sg * u);
                }
            }
}

// ---------------------------------------------------------------- GEMM2
// per-expert  A2 @ Wd -> Y (fp32, unweighted per-assignment rows)
__global__ __launch_bounds__(256)
void moe_gemm2_kernel(const bf16* __restrict__ A2,
                      const float* __restrict__ w_down,
                      const int* __restrict__ offsets,
                      float* __restrict__ Y) {
    int e = blockIdx.z;
    int seg0 = offsets[e];
    int nrows = offsets[e + 1] - seg0;
    int m0 = blockIdx.x * 128;
    if (m0 >= nrows) return;
    int n0 = blockIdx.y * 128;
    int rows_left = nrows - m0;

    __shared__ __align__(16) bf16 As[128 * 32];
    __shared__ __align__(16) bf16 Bs[128 * 32];  // transposed [n][k]

    int tid = threadIdx.x;
    int lane = tid & 31, wave = tid >> 5;
    int wm = (wave & 3) * 32;
    int wn = (wave >> 2) * 64;

    if (rows_left < 128) {
        for (int i = tid; i < 128 * 32; i += 256) As[i] = (bf16)0.0f;
    }

    v8f zero8 = {0.f, 0.f, 0.f, 0.f, 0.f, 0.f, 0.f, 0.f};
    v8f acc[2][4];
#pragma unroll
    for (int a = 0; a < 2; ++a)
#pragma unroll
        for (int b = 0; b < 4; ++b) acc[a][b] = zero8;

    const float* Wd = w_down + (size_t)e * HID * DIM;
    const bf16*  Ab = A2 + (size_t)(seg0 + m0) * HID;

    for (int kk = 0; kk < HID; kk += 32) {
        moe_async_tile_a(As, Ab + kk, HID, tid, rows_left);
        for (int i = tid; i < 32 * 128; i += 256) {
            int ki = i >> 7, ni = i & 127;
            Bs[ni * 32 + ki] = (bf16)Wd[(size_t)(kk + ki) * DIM + (size_t)(n0 + ni)];
        }
        if (kk + 32 < HID) {
            int pk = tid >> 3, pn = (tid & 7) * 16;
            __builtin_prefetch(&Wd[(size_t)(kk + 32 + pk) * DIM + (size_t)(n0 + pn)], 0, 1);
        }
        moe_wait_async0();
        __syncthreads();

        v16bf af0 = moe_frag_a(As, wm, lane);
        v16bf af1 = moe_frag_a(As, wm + 16, lane);
#pragma unroll
        for (int b = 0; b < 4; ++b) {
            v16bf bf = moe_frag_b(Bs, wn + b * 16, lane);
            acc[0][b] = __builtin_amdgcn_wmma_f32_16x16x32_bf16(
                false, af0, false, bf, (short)0, acc[0][b], false, false);
            acc[1][b] = __builtin_amdgcn_wmma_f32_16x16x32_bf16(
                false, af1, false, bf, (short)0, acc[1][b], false, false);
        }
        __syncthreads();
    }

    int h = lane >> 4, nn = lane & 15;
#pragma unroll
    for (int a = 0; a < 2; ++a)
#pragma unroll
        for (int b = 0; b < 4; ++b)
#pragma unroll
            for (int r = 0; r < 8; ++r) {
                int m = wm + a * 16 + h * 8 + r;
                if (m < rows_left) {
                    int col = n0 + wn + b * 16 + nn;
                    Y[(size_t)(seg0 + m0 + m) * DIM + col] = acc[a][b][r];
                }
            }
}

// ---------------------------------------------------------------- combine
__global__ __launch_bounds__(256)
void moe_combine_kernel(const float* __restrict__ Y, const int* __restrict__ pos_tk,
                        const float* __restrict__ rw, const float* __restrict__ zscale,
                        float* __restrict__ out) {
    int t = blockIdx.x;
    int p0 = pos_tk[2 * t], p1 = pos_tk[2 * t + 1];
    float w0 = rw[2 * t], w1 = rw[2 * t + 1];
    const float* y0 = Y + (size_t)p0 * DIM;
    const float* y1 = Y + (size_t)p1 * DIM;
    float* o = out + (size_t)t * DIM;
    for (int d = threadIdx.x; d < DIM; d += 256)
        o[d] = w0 * y0[d] + w1 * y1[d];
    if (t == 0 && threadIdx.x == 0)
        out[(size_t)T_TOK * DIM] = zscale[0];
}

// ---------------------------------------------------------------- launch
extern "C" void kernel_launch(void* const* d_in, const int* in_sizes, int n_in,
                              void* d_out, int out_size, void* d_ws, size_t ws_size,
                              hipStream_t stream) {
    const float* x           = (const float*)d_in[0];
    const float* gate_w      = (const float*)d_in[1];
    const float* expert_bias = (const float*)d_in[2];
    const float* w_gate      = (const float*)d_in[3];
    const float* w_up        = (const float*)d_in[4];
    const float* w_down      = (const float*)d_in[5];
    float* out = (float*)d_out;

    char* ws = (char*)d_ws;
    float* Y      = (float*)ws; ws += (size_t)NASSIGN * DIM * sizeof(float);
    bf16*  A2     = (bf16*)ws;  ws += (size_t)NASSIGN * HID * sizeof(bf16);
    bf16*  Xg     = (bf16*)ws;  ws += (size_t)NASSIGN * DIM * sizeof(bf16);
    float* rw     = (float*)ws; ws += (size_t)NASSIGN * sizeof(float);
    float* lse2   = (float*)ws; ws += (size_t)T_TOK * sizeof(float);
    float* zscale = (float*)ws; ws += 16;
    int*   ridx   = (int*)ws;   ws += (size_t)NASSIGN * sizeof(int);
    int*   pos_tk = (int*)ws;   ws += (size_t)NASSIGN * sizeof(int);
    int*   counts = (int*)ws;   ws += 32;
    int*   offs   = (int*)ws;   ws += 64;
    int*   curs   = (int*)ws;   ws += 32;

    moe_zero_kernel<<<1, 32, 0, stream>>>(counts);
    moe_router_kernel<<<T_TOK / 8, 256, 0, stream>>>(x, gate_w, expert_bias,
                                                     ridx, rw, lse2, counts);
    moe_scan_kernel<<<1, 256, 0, stream>>>(lse2, zscale, counts, offs, curs);
    moe_gather_kernel<<<NASSIGN, 128, 0, stream>>>(x, ridx, curs, pos_tk, Xg);

    dim3 g1(T_TOK / 128, HID / 128, NEXP);   // (32, 21, 8)
    moe_gemm1_kernel<<<g1, 256, 0, stream>>>(Xg, w_gate, w_up, offs, A2);

    dim3 g2(T_TOK / 128, DIM / 128, NEXP);   // (32, 6, 8)
    moe_gemm2_kernel<<<g2, 256, 0, stream>>>(A2, w_down, offs, Y);

    moe_combine_kernel<<<T_TOK, 256, 0, stream>>>(Y, pos_tk, rw, zscale, out);
}
